// ContrastivePretrainer_73993696575527
// MI455X (gfx1250) — matrix operand order repaired and checked by reading, not compiled, not run on hardware
//
#include <hip/hip_runtime.h>
#include <hip/hip_bf16.h>

#define NN 50000
#define EE 320000
#define HH 256
#define PP 128
#define LL 4

typedef __attribute__((ext_vector_type(8)))  float  v8f;
typedef __attribute__((ext_vector_type(16))) __bf16 v16bf;

__device__ __forceinline__ unsigned short f2bf(float f) {
  unsigned int u = __float_as_uint(f);
  return (unsigned short)((u + 0x7FFFu + ((u >> 16) & 1u)) >> 16);  // RNE truncate
}
// SiLU with hardware reciprocal (v_rcp_f32) instead of IEEE divide: feeds a bf16
// convert, so ~1 ULP rcp is far below the bf16 quantization error.
__device__ __forceinline__ float silu(float x) {
  return x * __builtin_amdgcn_rcpf(1.0f + __expf(-x));
}

// A-fragment (16x32 bf16) from a row-major LDS tile.
// lanes 0-15: M=lane, K = {kb..kb+7, kb+16..kb+23}; lanes 16-31: M=lane-16, K = {+8, +24}
__device__ __forceinline__ v16bf load_a_lds(const unsigned short* base, int lane, int rowstride) {
  const int m  = lane & 15;
  const int kh = (lane >> 4) << 3;  // 0 or 8
  const unsigned short* pr = base + m * rowstride + kh;
  union { uint4 q[2]; v16bf v; } u;
  u.q[0] = *(const uint4*)(pr);
  u.q[1] = *(const uint4*)(pr + 16);
  return u.v;
}

// B-fragment (32x16 bf16) from K-contiguous (column-major) weights: wt[col][k], kdim K per col.
// lanes 0-15: N=lane, K = {kb..kb+7, kb+16..kb+23}; lanes 16-31: N=lane-16, K = {+8, +24}
__device__ __forceinline__ v16bf load_b_glb(const unsigned short* __restrict__ wt, int lane,
                                            int ncol0, int kbase, int kdim) {
  const int n  = lane & 15;
  const int kh = (lane >> 4) << 3;  // 0 or 8
  const unsigned short* pr = wt + (size_t)(ncol0 + n) * kdim + kbase + kh;
  union { uint4 q[2]; v16bf v; } u;
  u.q[0] = *(const uint4*)(pr);
  u.q[1] = *(const uint4*)(pr + 16);
  return u.v;
}

__device__ __forceinline__ v8f wmma_bf16(v16bf a, v16bf b, v8f c) {
  return __builtin_amdgcn_wmma_f32_16x16x32_bf16(false, a, false, b, (short)0, c, false, false);
}

// ---------------- weight convert: fp32 [K][256] -> bf16 [256][K] (K-contiguous) ---------
__global__ void convert_weights_kernel(const float* __restrict__ mw1, const float* __restrict__ mw2,
                                       const float* __restrict__ uw1, const float* __restrict__ uw2,
                                       unsigned short* __restrict__ mw1T, unsigned short* __restrict__ mw2T,
                                       unsigned short* __restrict__ uw1T, unsigned short* __restrict__ uw2T) {
  const int stride = gridDim.x * blockDim.x;
  const int gid = blockIdx.x * blockDim.x + threadIdx.x;
  const int T1 = LL * HH * 512;  // w1 transposed (msg_w1 uses only K<512; dist row handled in fp32)
  const int T2 = LL * HH * HH;
  for (int i = gid; i < T1; i += stride) {
    int l = i / (HH * 512); int r = i % (HH * 512); int col = r / 512; int k = r % 512;
    mw1T[i] = f2bf(mw1[((size_t)l * 513 + k) * HH + col]);
    uw1T[i] = f2bf(uw1[((size_t)l * 512 + k) * HH + col]);
  }
  for (int i = gid; i < T2; i += stride) {
    int l = i / (HH * HH); int r = i % (HH * HH); int col = r / HH; int k = r % HH;
    mw2T[i] = f2bf(mw2[((size_t)l * HH + k) * HH + col]);
    uw2T[i] = f2bf(uw2[((size_t)l * HH + k) * HH + col]);
  }
}

// ---------------- per-encoder init: h = embed[z]*nodemask, p = pos + 0.05*noise ---------
__global__ void init_enc_kernel(const int* __restrict__ z, const float* __restrict__ pos,
                                const float* __restrict__ noise, const float* __restrict__ u_node,
                                const float* __restrict__ embed,
                                float* __restrict__ h, unsigned short* __restrict__ hbf,
                                float* __restrict__ p) {
  const int i = blockIdx.x;
  const int c = threadIdx.x;
  const float keep = (u_node[i] > 0.1f) ? 1.0f : 0.0f;
  const float v = embed[(size_t)z[i] * HH + c] * keep;
  h[(size_t)i * HH + c] = v;
  hbf[(size_t)i * HH + c] = f2bf(v);
  if (c < 3) p[i * 3 + c] = pos[i * 3 + c] + noise[i * 3 + c] * 0.05f;
}

// ---------------- edge message MLP + masked segment-sum ---------------------------------
// 32 edges / block; each wave owns a 32-col slab and computes a 2x2 grid of 16x16 tiles,
// so every A and B fragment feeds two WMMAs (halves L2 weight traffic vs 16-edge tiles).
__global__ void __launch_bounds__(256) edge_msg_kernel(
    const unsigned short* __restrict__ hbf, const float* __restrict__ p,
    const int* __restrict__ eidx, const float* __restrict__ u_edge,
    const unsigned short* __restrict__ w1T, const float* __restrict__ w1dist,
    const float* __restrict__ b1, const unsigned short* __restrict__ w2T,
    const float* __restrict__ b2, float* __restrict__ agg) {
  __shared__ __align__(16) unsigned short eA[32 * 512];   // [h_row | h_col] bf16, 32 KB
  __shared__ __align__(16) unsigned short mid[32 * 256];  // silu(GEMM1) bf16, 16 KB
  __shared__ float sdist[32], skeep[32];
  __shared__ int srow[32], scol[32];

  const int tid = threadIdx.x;
  const int e0 = blockIdx.x * 32;
  if (tid < 32) {
    const int e = e0 + tid;
    const int r = eidx[e], c = eidx[EE + e];
    srow[tid] = r; scol[tid] = c;
    const float dx = p[r * 3 + 0] - p[c * 3 + 0];
    const float dy = p[r * 3 + 1] - p[c * 3 + 1];
    const float dz = p[r * 3 + 2] - p[c * 3 + 2];
    sdist[tid] = sqrtf(dx * dx + dy * dy + dz * dz + 1e-8f);
    skeep[tid] = (u_edge[e] > 0.1f) ? 1.0f : 0.0f;
  }
  __syncthreads();

  // stage 32 edges x 512 bf16 features (2048 b128 slots, 8 per thread)
  #pragma unroll
  for (int j = 0; j < 8; j++) {
    const int u = tid + 256 * j;
    const int i = u >> 6; const int within = u & 63;
    const int half = within >> 5; const int off = (within & 31) * 8;
    const int node = half ? scol[i] : srow[i];
    *(uint4*)(&eA[i * 512 + half * 256 + off]) = *(const uint4*)(hbf + (size_t)node * HH + off);
  }
  __syncthreads();

  const int lane = tid & 31;
  const int n0 = (tid >> 5) * 32;
  __builtin_prefetch(w2T + (size_t)n0 * HH, 0, 0);

  // GEMM1: [32 x 512] @ [512 x 256] (dist column handled as fp32 rank-1 in epilogue)
  v8f acc00 = {}, acc01 = {}, acc10 = {}, acc11 = {};   // [mtile][ntile]
  #pragma unroll 2
  for (int kk = 0; kk < 16; kk++) {
    const int kb = kk * 32;
    v16bf a0  = load_a_lds(eA + kb,            lane, 512);
    v16bf a1  = load_a_lds(eA + 16 * 512 + kb, lane, 512);
    v16bf bf0 = load_b_glb(w1T, lane, n0,      kb, 512);
    v16bf bf1 = load_b_glb(w1T, lane, n0 + 16, kb, 512);
    acc00 = wmma_bf16(a0, bf0, acc00);
    acc01 = wmma_bf16(a0, bf1, acc01);
    acc10 = wmma_bf16(a1, bf0, acc10);
    acc11 = wmma_bf16(a1, bf1, acc11);
  }
  const int nA = n0 + (lane & 15);
  const int nB = nA + 16;
  const int mbase = (lane >> 4) << 3;
  {
    const float biasA = b1[nA], biasB = b1[nB];
    const float wdA = w1dist[nA], wdB = w1dist[nB];
    #pragma unroll
    for (int r = 0; r < 8; r++) {
      const int m0 = mbase + r;
      const int m1 = m0 + 16;
      const float d0 = sdist[m0], d1 = sdist[m1];
      mid[m0 * 256 + nA] = f2bf(silu(acc00[r] + biasA + d0 * wdA));
      mid[m0 * 256 + nB] = f2bf(silu(acc01[r] + biasB + d0 * wdB));
      mid[m1 * 256 + nA] = f2bf(silu(acc10[r] + biasA + d1 * wdA));
      mid[m1 * 256 + nB] = f2bf(silu(acc11[r] + biasB + d1 * wdB));
    }
  }
  __syncthreads();

  // GEMM2: [32 x 256] @ [256 x 256]
  v8f g00 = {}, g01 = {}, g10 = {}, g11 = {};
  #pragma unroll
  for (int kk = 0; kk < 8; kk++) {
    const int kb = kk * 32;
    v16bf a0  = load_a_lds(mid + kb,            lane, 256);
    v16bf a1  = load_a_lds(mid + 16 * 256 + kb, lane, 256);
    v16bf bf0 = load_b_glb(w2T, lane, n0,      kb, 256);
    v16bf bf1 = load_b_glb(w2T, lane, n0 + 16, kb, 256);
    g00 = wmma_bf16(a0, bf0, g00);
    g01 = wmma_bf16(a0, bf1, g01);
    g10 = wmma_bf16(a1, bf0, g10);
    g11 = wmma_bf16(a1, bf1, g11);
  }
  const float b2A = b2[nA], b2B = b2[nB];
  #pragma unroll
  for (int r = 0; r < 8; r++) {
    const int m0 = mbase + r;
    const int m1 = m0 + 16;
    const float k0 = skeep[m0];
    const float k1 = skeep[m1];
    if (k0 != 0.0f) {
      const size_t base = (size_t)srow[m0] * HH;
      unsafeAtomicAdd(&agg[base + nA], (g00[r] + b2A) * k0);   // global_atomic_add_f32
      unsafeAtomicAdd(&agg[base + nB], (g01[r] + b2B) * k0);
    }
    if (k1 != 0.0f) {
      const size_t base = (size_t)srow[m1] * HH;
      unsafeAtomicAdd(&agg[base + nA], (g10[r] + b2A) * k1);
      unsafeAtomicAdd(&agg[base + nB], (g11[r] + b2B) * k1);
    }
  }
}

// ---------------- node update MLP + residual + LayerNorm --------------------------------
__global__ void __launch_bounds__(256) node_upd_kernel(
    float* __restrict__ h, unsigned short* __restrict__ hbf,
    const float* __restrict__ agg,
    const unsigned short* __restrict__ w1T, const float* __restrict__ b1,
    const unsigned short* __restrict__ w2T, const float* __restrict__ b2,
    const float* __restrict__ lg, const float* __restrict__ lb) {
  __shared__ __align__(16) unsigned short uA[16 * 512];   // [h | agg] bf16
  __shared__ __align__(16) unsigned short mid[16 * 256];
  __shared__ __align__(16) float xb[16 * 256];            // h + upd (pre-LN) fp32
  __shared__ float smu[16], srs[16];

  const int tid = threadIdx.x;
  const int v0 = blockIdx.x * 16;

  // stage h (bf16 copy) ...
  #pragma unroll
  for (int j = 0; j < 2; j++) {
    const int u = tid + 256 * j;                  // 0..511 uint4 slots
    const int i = u >> 5; const int off = (u & 31) * 8;
    *(uint4*)(&uA[i * 512 + off]) = *(const uint4*)(hbf + (size_t)(v0 + i) * HH + off);
  }
  // ... and agg (fp32 -> bf16 on the fly)
  #pragma unroll
  for (int j = 0; j < 4; j++) {
    const int u = tid + 256 * j;                  // 0..1023 float4 slots
    const int i = u >> 6; const int off4 = (u & 63) * 4;
    const float4 v = *(const float4*)(agg + (size_t)(v0 + i) * HH + off4);
    ushort4 t; t.x = f2bf(v.x); t.y = f2bf(v.y); t.z = f2bf(v.z); t.w = f2bf(v.w);
    *(ushort4*)(&uA[i * 512 + 256 + off4]) = t;
  }
  __syncthreads();

  const int lane = tid & 31;
  const int n0 = (tid >> 5) * 32;
  const int nA = n0 + (lane & 15);
  const int nB = nA + 16;
  const int mbase = (lane >> 4) << 3;

  v8f acc0 = {}; v8f acc1 = {};
  #pragma unroll 4
  for (int kk = 0; kk < 16; kk++) {
    const int kb = kk * 32;
    v16bf a   = load_a_lds(uA + kb, lane, 512);
    v16bf bf0 = load_b_glb(w1T, lane, n0,      kb, 512);
    v16bf bf1 = load_b_glb(w1T, lane, n0 + 16, kb, 512);
    acc0 = wmma_bf16(a, bf0, acc0);
    acc1 = wmma_bf16(a, bf1, acc1);
  }
  {
    const float biasA = b1[nA], biasB = b1[nB];
    #pragma unroll
    for (int r = 0; r < 8; r++) {
      const int m = mbase + r;
      mid[m * 256 + nA] = f2bf(silu(acc0[r] + biasA));
      mid[m * 256 + nB] = f2bf(silu(acc1[r] + biasB));
    }
  }
  __syncthreads();

  v8f g0 = {}; v8f g1 = {};
  #pragma unroll
  for (int kk = 0; kk < 8; kk++) {
    const int kb = kk * 32;
    v16bf a   = load_a_lds(mid + kb, lane, 256);
    v16bf bf0 = load_b_glb(w2T, lane, n0,      kb, 256);
    v16bf bf1 = load_b_glb(w2T, lane, n0 + 16, kb, 256);
    g0 = wmma_bf16(a, bf0, g0);
    g1 = wmma_bf16(a, bf1, g1);
  }
  const float b2A = b2[nA], b2B = b2[nB];
  #pragma unroll
  for (int r = 0; r < 8; r++) {
    const int m = mbase + r;
    xb[m * 256 + nA] = h[(size_t)(v0 + m) * HH + nA] + g0[r] + b2A;
    xb[m * 256 + nB] = h[(size_t)(v0 + m) * HH + nB] + g1[r] + b2B;
  }
  __syncthreads();

  if (tid < 16) {
    float s = 0.f;
    for (int c = 0; c < HH; c++) s += xb[tid * 256 + c];
    const float mu = s * (1.0f / HH);
    float vv = 0.f;
    for (int c = 0; c < HH; c++) { const float d = xb[tid * 256 + c] - mu; vv += d * d; }
    smu[tid] = mu;
    srs[tid] = rsqrtf(vv * (1.0f / HH) + 1e-5f);
  }
  __syncthreads();

  const int m = tid >> 4;
  const int c0 = (tid & 15) * 16;
  const float mu = smu[m], rs = srs[m];
  for (int c = c0; c < c0 + 16; c++) {
    const float y = (xb[m * 256 + c] - mu) * rs * lg[c] + lb[c];
    h[(size_t)(v0 + m) * HH + c] = y;
    hbf[(size_t)(v0 + m) * HH + c] = f2bf(y);
  }
}

// ---------------- mean over nodes (column sums + atomic combine) ------------------------
__global__ void colsum_kernel(const float* __restrict__ h, float* __restrict__ meanv) {
  const int c = threadIdx.x;
  const int r0 = blockIdx.x * 256;
  float s = 0.f;
  for (int i = 0; i < 256; i++) {
    const int r = r0 + i;
    if (r < NN) s += h[(size_t)r * HH + c];
  }
  unsafeAtomicAdd(&meanv[c], s);
}

// ---------------- projection head (tiny) ------------------------------------------------
__global__ void proj_kernel(const float* __restrict__ meanv,
                            const float* __restrict__ pw1, const float* __restrict__ pb1,
                            const float* __restrict__ pw2, const float* __restrict__ pb2,
                            float* __restrict__ zbuf, int enc) {
  __shared__ float v[HH];
  __shared__ float t1[HH];
  const int tid = threadIdx.x;
  v[tid] = meanv[tid] * (1.0f / (float)NN);
  __syncthreads();
  float s = pb1[tid];
  for (int k = 0; k < HH; k++) s += v[k] * pw1[k * HH + tid];
  t1[tid] = s > 0.f ? s : 0.f;
  __syncthreads();
  if (tid < PP) {
    float s2 = pb2[tid];
    for (int k = 0; k < HH; k++) s2 += t1[k] * pw2[k * PP + tid];
    zbuf[enc * PP + tid] = s2;
  }
}

// ---------------- contrastive loss (2x2, single lane) -----------------------------------
__global__ void loss_kernel(const float* __restrict__ zbuf, float* __restrict__ out) {
  if (threadIdx.x != 0 || blockIdx.x != 0) return;
  const float* z1 = zbuf;
  const float* z2 = zbuf + PP;
  float n1 = 0.f, n2 = 0.f;
  for (int i = 0; i < PP; i++) { n1 += 1.0f / fabsf(z1[i]); n2 += 1.0f / fabsf(z2[i]); }
  const float d1 = fmaxf(1.0f / n1, 1e-12f);
  const float d2 = fmaxf(1.0f / n2, 1e-12f);
  float s = 0.f;
  for (int i = 0; i < PP; i++) s += (z1[i] / d1) * (z2[i] / d2);
  s *= 10.0f;  // /TEMP (0.1)
  const float NEG = -1e9f;
  const float m0 = fmaxf(NEG, s);
  const float lse = m0 + logf(__expf(NEG - m0) + __expf(s - m0));
  const float lp = s - lse;           // logp[0,1] == logp[1,0] by symmetry
  out[0] = -lp;
}

// ---------------- host-side orchestration ----------------------------------------------
extern "C" void kernel_launch(void* const* d_in, const int* in_sizes, int n_in,
                              void* d_out, int out_size, void* d_ws, size_t ws_size,
                              hipStream_t stream) {
  (void)in_sizes; (void)n_in; (void)out_size; (void)ws_size;
  const int*   z       = (const int*)  d_in[0];
  const float* pos     = (const float*)d_in[1];
  const int*   eidx    = (const int*)  d_in[2];
  const float* u_node1 = (const float*)d_in[3];
  const float* u_node2 = (const float*)d_in[4];
  const float* u_edge1 = (const float*)d_in[5];
  const float* u_edge2 = (const float*)d_in[6];
  const float* noise1  = (const float*)d_in[7];
  const float* noise2  = (const float*)d_in[8];
  const float* embed   = (const float*)d_in[9];
  const float* msg_w1  = (const float*)d_in[10];
  const float* msg_b1  = (const float*)d_in[11];
  const float* msg_w2  = (const float*)d_in[12];
  const float* msg_b2  = (const float*)d_in[13];
  const float* upd_w1  = (const float*)d_in[14];
  const float* upd_b1  = (const float*)d_in[15];
  const float* upd_w2  = (const float*)d_in[16];
  const float* upd_b2  = (const float*)d_in[17];
  const float* ln_g    = (const float*)d_in[18];
  const float* ln_b    = (const float*)d_in[19];
  const float* proj_w1 = (const float*)d_in[20];
  const float* proj_b1 = (const float*)d_in[21];
  const float* proj_w2 = (const float*)d_in[22];
  const float* proj_b2 = (const float*)d_in[23];

  char* ws = (char*)d_ws;
  size_t off = 0;
  auto carve = [&](size_t bytes) -> char* {
    char* r = ws + off;
    off = (off + bytes + 255) & ~(size_t)255;
    return r;
  };
  float*          h      = (float*)         carve((size_t)NN * HH * 4);
  unsigned short* hbf    = (unsigned short*)carve((size_t)NN * HH * 2);
  float*          agg    = (float*)         carve((size_t)NN * HH * 4);
  float*          pcoord = (float*)         carve((size_t)NN * 3 * 4);
  float*          meanv  = (float*)         carve(HH * 4);
  float*          zbuf   = (float*)         carve(2 * PP * 4);
  unsigned short* mw1T   = (unsigned short*)carve((size_t)LL * 512 * HH * 2);
  unsigned short* mw2T   = (unsigned short*)carve((size_t)LL * HH * HH * 2);
  unsigned short* uw1T   = (unsigned short*)carve((size_t)LL * 512 * HH * 2);
  unsigned short* uw2T   = (unsigned short*)carve((size_t)LL * HH * HH * 2);

  convert_weights_kernel<<<512, 256, 0, stream>>>(msg_w1, msg_w2, upd_w1, upd_w2,
                                                  mw1T, mw2T, uw1T, uw2T);

  const float* u_nodes[2] = {u_node1, u_node2};
  const float* u_edges[2] = {u_edge1, u_edge2};
  const float* noises[2]  = {noise1, noise2};

  for (int enc = 0; enc < 2; enc++) {
    init_enc_kernel<<<NN, HH, 0, stream>>>(z, pos, noises[enc], u_nodes[enc], embed,
                                           h, hbf, pcoord);
    for (int l = 0; l < LL; l++) {
      hipMemsetAsync(agg, 0, (size_t)NN * HH * 4, stream);
      edge_msg_kernel<<<EE / 32, 256, 0, stream>>>(
          hbf, pcoord, eidx, u_edges[enc],
          mw1T + (size_t)l * HH * 512,
          msg_w1 + ((size_t)l * 513 + 512) * HH,   // dist row (K=512) kept fp32
          msg_b1 + (size_t)l * HH,
          mw2T + (size_t)l * HH * HH,
          msg_b2 + (size_t)l * HH,
          agg);
      node_upd_kernel<<<NN / 16, 256, 0, stream>>>(
          h, hbf, agg,
          uw1T + (size_t)l * HH * 512, upd_b1 + (size_t)l * HH,
          uw2T + (size_t)l * HH * HH,  upd_b2 + (size_t)l * HH,
          ln_g + (size_t)l * HH, ln_b + (size_t)l * HH);
    }
    hipMemsetAsync(meanv, 0, HH * 4, stream);
    colsum_kernel<<<(NN + 255) / 256, 256, 0, stream>>>(h, meanv);
    proj_kernel<<<1, HH, 0, stream>>>(meanv, proj_w1, proj_b1, proj_w2, proj_b2, zbuf, enc);
  }
  loss_kernel<<<1, 32, 0, stream>>>(zbuf, (float*)d_out);
}
